// IdScaleAttn_55121610277072
// MI455X (gfx1250) — matrix operand order repaired and checked by reading, not compiled, not run on hardware
//
#include <hip/hip_runtime.h>
#include <hip/hip_fp16.h>
#include <math.h>

typedef _Float16 v16h __attribute__((ext_vector_type(16)));
typedef _Float16 h8   __attribute__((ext_vector_type(8)));
typedef float    v8f  __attribute__((ext_vector_type(8)));
typedef float    f4v  __attribute__((ext_vector_type(4)));

#define FEAT 256
#define HEADS 8
#define MAPS 5

// Async DMA: global memory -> LDS, 16 bytes per enabled lane. ASYNCcnt-tracked.
__device__ __forceinline__ void async_copy16(unsigned lds_off, const void* gptr) {
  asm volatile("global_load_async_to_lds_b128 %0, %1, off"
               :: "v"(lds_off), "v"(gptr) : "memory");
}
__device__ __forceinline__ void wait_async0() {
  asm volatile("s_wait_asynccnt 0" ::: "memory");
}

// ---------------------------------------------------------------------------
// GEMM: C[M x 256] = A[M x 256] @ B^T + bias, B row-major [256 x 256] as
// (n, k) -- exactly val_w / out_w from the reference (out = x @ W.T).
// A may be a concatenation of two row blocks (A1 rows1 rows, then A2).
// Tiles staged fp32 in LDS via async DMA (double-buffered); converted to f16
// at fragment build (v_cvt co-executes with XDL WMMA); f32 accumulate.
// Block: 256 threads = 8 waves. Tile M=128 (wave owns 16-row strip), N=64
// (4 n-subtiles of 16 per wave), K-step 32, 8 k-steps.
// ---------------------------------------------------------------------------
__global__ __launch_bounds__(256) void gemm256(
    const float* __restrict__ A1, int rows1,
    const float* __restrict__ A2, int Mtotal,
    const float* __restrict__ B,  const float* __restrict__ bias,
    _Float16* __restrict__ Ch, float* __restrict__ Cf)
{
  __shared__ __align__(16) float Albuf[2][128 * 40]; // rows padded to 40 f32
  __shared__ __align__(16) float Blbuf[2][64 * 40];

  const int t      = threadIdx.x;
  const int lane   = t & 31;
  const int wv     = t >> 5;
  const int blockM = blockIdx.x;
  const int blockN = blockIdx.y;

  const unsigned aBase[2] = { (unsigned)(uintptr_t)&Albuf[0][0],
                              (unsigned)(uintptr_t)&Albuf[1][0] };
  const unsigned bBase[2] = { (unsigned)(uintptr_t)&Blbuf[0][0],
                              (unsigned)(uintptr_t)&Blbuf[1][0] };

  // Issue async copies for k-step `ks` into buffer `bi`.
  auto load_tiles = [&](int bi, int ks) {
    // A tile: 128 rows x 32 f32 = 1024 16B chunks, 4 per thread
    #pragma unroll
    for (int it = 0; it < 4; ++it) {
      int idx = t + it * 256;
      int r = idx >> 3, c4 = idx & 7;
      int gm = blockM * 128 + r;
      if (gm < Mtotal) {
        const float* src = ((gm < rows1) ? (A1 + (size_t)gm * FEAT)
                                         : (A2 + (size_t)(gm - rows1) * FEAT))
                           + ks * 32 + c4 * 4;
        async_copy16(aBase[bi] + (unsigned)(r * 160 + c4 * 16), src);
      } else { // zero-fill OOB rows (only the one partial M-block)
        f4v z = {0.f, 0.f, 0.f, 0.f};
        *(f4v*)&Albuf[bi][r * 40 + c4 * 4] = z;
      }
    }
    // B tile: 64 rows x 32 f32 = 512 16B chunks, 2 per thread
    #pragma unroll
    for (int it = 0; it < 2; ++it) {
      int idx = t + it * 256;
      int r = idx >> 3, c4 = idx & 7;
      int gn = blockN * 64 + r;
      const float* src = B + (size_t)gn * FEAT + ks * 32 + c4 * 4;
      async_copy16(bBase[bi] + (unsigned)(r * 160 + c4 * 16), src);
    }
  };

  v8f acc[4];
  #pragma unroll
  for (int i = 0; i < 4; ++i)
    #pragma unroll
    for (int r = 0; r < 8; ++r) acc[i][r] = 0.0f;

  load_tiles(0, 0);
  wait_async0();
  __syncthreads();

  for (int ks = 0; ks < 8; ++ks) {
    const int cur = ks & 1;
    if (ks < 7) load_tiles(cur ^ 1, ks + 1); // overlap DMA with WMMA

    // ---- A fragment (16x32 f16): lane<16 -> K {0..7,16..23}, lane>=16 -> +8
    const int mrow  = wv * 16 + (lane & 15);
    const int koffA = (lane >> 4) * 8;
    const float* Ab = &Albuf[cur][mrow * 40];
    f4v a00 = *(const f4v*)(Ab + koffA);
    f4v a01 = *(const f4v*)(Ab + koffA + 4);
    f4v a10 = *(const f4v*)(Ab + 16 + koffA);
    f4v a11 = *(const f4v*)(Ab + 20 + koffA);
    v16h afrag;
    #pragma unroll
    for (int i = 0; i < 4; ++i) {
      afrag[i]      = (_Float16)a00[i];
      afrag[4 + i]  = (_Float16)a01[i];
      afrag[8 + i]  = (_Float16)a10[i];
      afrag[12 + i] = (_Float16)a11[i];
    }

    #pragma unroll
    for (int nt = 0; nt < 4; ++nt) {
      // ---- B fragment (32x16 f16): lane<16 -> K {0..7,16..23}, lane>=16 -> +8
      const int nrow  = nt * 16 + (lane & 15);
      const int koffB = (lane >> 4) * 8;
      const float* Bb = &Blbuf[cur][nrow * 40 + koffB];
      f4v b0 = *(const f4v*)(Bb);
      f4v b1 = *(const f4v*)(Bb + 4);
      f4v b2 = *(const f4v*)(Bb + 16);
      f4v b3 = *(const f4v*)(Bb + 20);
      v16h bfrag;
      #pragma unroll
      for (int i = 0; i < 4; ++i) {
        bfrag[i]      = (_Float16)b0[i];
        bfrag[4 + i]  = (_Float16)b1[i];
        bfrag[8 + i]  = (_Float16)b2[i];
        bfrag[12 + i] = (_Float16)b3[i];
      }
      acc[nt] = __builtin_amdgcn_wmma_f32_16x16x32_f16(
          false, afrag, false, bfrag, (short)0, acc[nt], false, false);
    }

    if (ks < 7) {
      wait_async0();    // this wave's DMA for ks+1 done
      __syncthreads();  // all waves' DMA visible; all reads of next buf done
    }
  }

  // ---- epilogue: C/D layout -> lane holds (m = (lane>>4)*8 + r, n = lane&15)
  #pragma unroll
  for (int nt = 0; nt < 4; ++nt) {
    int gn = blockN * 64 + nt * 16 + (lane & 15);
    float bv = bias[gn];
    #pragma unroll
    for (int r = 0; r < 8; ++r) {
      int gm = blockM * 128 + wv * 16 + (lane >> 4) * 8 + r;
      if (gm < Mtotal) {
        float v = acc[nt][r] + bv;
        if (Ch) Ch[(size_t)gm * FEAT + gn] = (_Float16)v;
        else    Cf[(size_t)gm * FEAT + gn] = v;
      }
    }
  }
}

// ---------------------------------------------------------------------------
// Per-query: attention logits + softmax + bilinear weights + id gathers +
// weighted accumulation of pre-projected value rows (val_all, f16).
// One wave32 per query; 8 waves per block. Writes out_pre (f32).
// ---------------------------------------------------------------------------
__global__ __launch_bounds__(256) void attn_gather(
    const float* __restrict__ in_act,
    const int* __restrict__ bids, const int* __restrict__ mids,
    const int* __restrict__ xyids, const int* __restrict__ map_shapes,
    const float* __restrict__ scale_embed,
    const float* __restrict__ attn_w, const float* __restrict__ attn_b,
    const int* __restrict__ id0, const int* __restrict__ id1,
    const int* __restrict__ id2, const int* __restrict__ id3,
    const int* __restrict__ id4,
    const _Float16* __restrict__ valall,
    float* __restrict__ out_pre, int nQ)
{
  __shared__ __align__(16) float sAttnW[40 * 260]; // rows padded (16B aligned)
  __shared__ __align__(16) float sScale[MAPS * FEAT];
  __shared__ float sAttnB[40];
  __shared__ int   sMS[10];
  __shared__ __align__(16) float sAct[8 * FEAT]; // per-wave scratch
  __shared__ float sAw[8 * 48];

  const int t = threadIdx.x;
  for (int idx = t; idx < 40 * 256; idx += 256)
    sAttnW[(idx >> 8) * 260 + (idx & 255)] = attn_w[idx];
  for (int idx = t; idx < MAPS * FEAT; idx += 256) sScale[idx] = scale_embed[idx];
  if (t < 40) sAttnB[t] = attn_b[t];
  if (t < 10) sMS[t]    = map_shapes[t];
  __syncthreads();

  const int wv = t >> 5, lane = t & 31;
  const int n = blockIdx.x * 8 + wv;
  if (n >= nQ) return;

  const int mid = mids[n];
  const int b   = bids[n];

  // act = in_act + scale_embed[mid]; lane covers feats [lane*8, lane*8+8)
  float4 x0 = *(const float4*)(in_act + (size_t)n * FEAT + lane * 8);
  float4 x1 = *(const float4*)(in_act + (size_t)n * FEAT + lane * 8 + 4);
  float4 s0 = *(const float4*)&sScale[mid * FEAT + lane * 8];
  float4 s1 = *(const float4*)&sScale[mid * FEAT + lane * 8 + 4];
  float4 a0 = make_float4(x0.x + s0.x, x0.y + s0.y, x0.z + s0.z, x0.w + s0.w);
  float4 a1 = make_float4(x1.x + s1.x, x1.y + s1.y, x1.z + s1.z, x1.w + s1.w);
  *(float4*)&sAct[wv * FEAT + lane * 8]     = a0;
  *(float4*)&sAct[wv * FEAT + lane * 8 + 4] = a1;
  __builtin_amdgcn_wave_barrier(); // same-wave LDS ops are in order

  // attention logits: lane computes row j0=lane; lanes 0..7 also j1=32+lane
  const int j0 = lane, j1 = 32 + (lane & 7);
  float d0 = 0.f, d1 = 0.f;
  for (int k = 0; k < FEAT; k += 4) {
    float4 av = *(const float4*)&sAct[wv * FEAT + k]; // broadcast read
    float4 w0 = *(const float4*)&sAttnW[j0 * 260 + k];
    float4 w1 = *(const float4*)&sAttnW[j1 * 260 + k];
    d0 += av.x * w0.x + av.y * w0.y + av.z * w0.z + av.w * w0.w;
    d1 += av.x * w1.x + av.y * w1.y + av.z * w1.z + av.w * w1.w;
  }
  sAw[wv * 48 + j0] = d0 + sAttnB[j0];
  if (lane < 8) sAw[wv * 48 + j1] = d1 + sAttnB[j1];
  __builtin_amdgcn_wave_barrier();

  // softmax over maps for this lane's head (head = lane>>2, since each lane
  // accumulates feats [lane*8, lane*8+8) and head dim = 32)
  const int h5 = lane >> 2;
  float awv[MAPS], pm[MAPS];
  #pragma unroll
  for (int m = 0; m < MAPS; ++m) awv[m] = sAw[wv * 48 + h5 * MAPS + m];
  float mx = awv[0];
  #pragma unroll
  for (int m = 1; m < MAPS; ++m) mx = fmaxf(mx, awv[m]);
  float sum = 0.f;
  #pragma unroll
  for (int m = 0; m < MAPS; ++m) { pm[m] = __expf(awv[m] - mx); sum += pm[m]; }
  float inv = 1.0f / sum;
  #pragma unroll
  for (int m = 0; m < MAPS; ++m) pm[m] *= inv;

  // bilinear geometry: lanes 0..19 own sample s = m*4 + corner
  const int Hmid = sMS[2 * mid], Wmid = sMS[2 * mid + 1];
  const int xi = xyids[2 * n], yi = xyids[2 * n + 1];
  const float nx = ((float)xi + 0.5f) / (float)Wmid;
  const float ny = ((float)yi + 0.5f) / (float)Hmid;
  int s = lane;
  int m = s >> 2; if (m > 4) m = 4;
  const int c = s & 3;
  const int Hm = sMS[2 * m], Wm = sMS[2 * m + 1];
  const float sx = nx * (float)Wm - 0.5f;
  const float sy = ny * (float)Hm - 0.5f;
  const int cx = (int)floorf(sx) + (c & 1);
  const int cy = (int)floorf(sy) + (c >> 1);
  float wgt = (1.0f - fabsf(sx - (float)cx)) * (1.0f - fabsf(sy - (float)cy));
  if (s >= MAPS * 4) wgt = 0.0f;
  const int xcl = cx < 0 ? 0 : (cx > Wm - 1 ? Wm - 1 : cx);
  const int ycl = cy < 0 ? 0 : (cy > Hm - 1 ? Hm - 1 : cy);
  const int* idm = (m == 0) ? id0 : (m == 1) ? id1 : (m == 2) ? id2
                 : (m == 3) ? id3 : id4;
  const int fid = idm[((size_t)b * Hm + ycl) * (size_t)Wm + xcl];

  // gather pre-projected value rows; lane accumulates its 8 feats
  float acc[8];
  #pragma unroll
  for (int i = 0; i < 8; ++i) acc[i] = 0.f;
  #pragma unroll
  for (int s2 = 0; s2 < MAPS * 4; ++s2) {
    int   fs = __shfl(fid, s2, 32);
    float sw = __shfl(wgt, s2, 32);
    float fw = pm[s2 >> 2] * sw; // this lane's head weight for map s2>>2
    h8 row = *(const h8*)(valall + (size_t)fs * FEAT + lane * 8);
    #pragma unroll
    for (int i = 0; i < 8; ++i) acc[i] += fw * (float)row[i];
  }

  float4 o0 = make_float4(acc[0], acc[1], acc[2], acc[3]);
  float4 o1 = make_float4(acc[4], acc[5], acc[6], acc[7]);
  *(float4*)(out_pre + (size_t)n * FEAT + lane * 8)     = o0;
  *(float4*)(out_pre + (size_t)n * FEAT + lane * 8 + 4) = o1;
}

// ---------------------------------------------------------------------------
extern "C" void kernel_launch(void* const* d_in, const int* in_sizes, int n_in,
                              void* d_out, int out_size, void* d_ws, size_t ws_size,
                              hipStream_t stream) {
  const float* in_act  = (const float*)d_in[0];
  const float* pas     = (const float*)d_in[1];
  const int*   bids    = (const int*)d_in[2];
  const int*   mids    = (const int*)d_in[3];
  const int*   xy      = (const int*)d_in[4];
  const int*   mshapes = (const int*)d_in[5];
  const float* scale   = (const float*)d_in[6];
  const float* attn_w  = (const float*)d_in[7];
  const float* attn_b  = (const float*)d_in[8];
  const float* val_w   = (const float*)d_in[9];
  const float* val_b   = (const float*)d_in[10];
  const float* out_w   = (const float*)d_in[11];
  const float* out_b   = (const float*)d_in[12];
  const int*   id0     = (const int*)d_in[13];
  const int*   id1     = (const int*)d_in[14];
  const int*   id2     = (const int*)d_in[15];
  const int*   id3     = (const int*)d_in[16];
  const int*   id4     = (const int*)d_in[17];

  const int nAct = in_sizes[0] / FEAT;
  const int nPas = in_sizes[1] / FEAT;
  const int M    = nAct + nPas;

  _Float16* valall  = (_Float16*)d_ws;
  float*    out_pre = (float*)((char*)d_ws + (size_t)M * FEAT * sizeof(_Float16));
  float*    outF    = (float*)d_out;

  // 1) val_all = concat(in_act, pas) @ val_w.T + val_b   (f16 output)
  dim3 g1((M + 127) / 128, 4);
  gemm256<<<g1, 256, 0, stream>>>(in_act, nAct, pas, M, val_w, val_b,
                                  valall, nullptr);

  // 2) attention + bilinear gather of pre-projected values -> out_pre (f32)
  attn_gather<<<dim3((nAct + 7) / 8), 256, 0, stream>>>(
      in_act, bids, mids, xy, mshapes, scale, attn_w, attn_b,
      id0, id1, id2, id3, id4, valall, out_pre, nAct);

  // 3) out = out_pre @ out_w.T + out_b   (f32 output)
  dim3 g3((nAct + 127) / 128, 4);
  gemm256<<<g3, 256, 0, stream>>>(out_pre, nAct, nullptr, nAct, out_w, out_b,
                                  nullptr, outF);
}
